// NaiveEngram_32186484916590
// MI455X (gfx1250) — matrix operand order repaired and checked by reading, not compiled, not run on hardware
//
#include <hip/hip_runtime.h>
#include <hip/hip_bf16.h>

// ---------------- problem constants ----------------
#define B_   4
#define S_   4096
#define D_   2048
#define NH_  24
#define V_   50000
#define HD_  64
#define EH_  1536          // NH_*HD_
#define M_   (B_*S_)       // 16384 rows

// GEMM tiling
#define BM 64
#define BN 64
#define LDA 72             // padded LDS row stride in bf16 elems (144 bytes)
#define TSZ (BM * LDA)     // elems per LDS tile (4608)

typedef __attribute__((ext_vector_type(16))) __bf16 v16bf;
typedef __attribute__((ext_vector_type(8)))  float  v8f;
typedef __attribute__((ext_vector_type(4)))  unsigned int v4u;
typedef __attribute__((ext_vector_type(8)))  int  v8i;
typedef __attribute__((ext_vector_type(4)))  int  v4i;

union FragBF { v16bf v; uint4 q[2]; };

__device__ __forceinline__ unsigned short f2bf(float f) {
    unsigned u = __float_as_uint(f);
    u += 0x7FFFu + ((u >> 16) & 1u);      // round-to-nearest-even
    return (unsigned short)(u >> 16);
}

// ---- Tensor Data Mover: 64x64 bf16 2D tile -> LDS with 16B row padding ----
// Tensor: [D_ rows x EH_ cols] bf16, row stride EH_. Tile: 64 rows x 64 elems.
// pad_enable: +4 DWORDs (16B) after every 32 DWORDs (128B = one tile row)
// => LDS row stride 144B == LDA bf16 elems, matching the WMMA fragment reads.
__device__ __forceinline__ void tdm_load_tile_b(const unsigned short* gtile,
                                                unsigned lds_byte_addr) {
    const unsigned long long ga = (unsigned long long)(size_t)gtile;
    v4u g0;
    g0.x = 1u;                                    // count=1 (user descriptor)
    g0.y = lds_byte_addr;                         // LDS dest (bytes)
    g0.z = (unsigned)(ga & 0xFFFFFFFFu);          // global_addr[31:0]
    g0.w = (unsigned)((ga >> 32) & 0x01FFFFFFu)   // global_addr[56:32]
         | (2u << 30);                            // type = 2 ("image")
    v8i g1;
    g1[0] = (int)((1u << 16)      // data_size = 1 -> 2 bytes
                | (1u << 20)      // pad_enable
                | (4u << 22)      // pad_interval code 4 -> every 32 DWORDs
                | (3u << 25));    // pad_amount  code 3 -> 4 DWORDs
    g1[1] = (int)(((unsigned)EH_ & 0xFFFFu) << 16);          // tensor_dim0 lo16
    g1[2] = (int)((((unsigned)EH_ >> 16) & 0xFFFFu)          // tensor_dim0 hi16
                | (((unsigned)D_ & 0xFFFFu) << 16));         // tensor_dim1 lo16
    g1[3] = (int)((((unsigned)D_ >> 16) & 0xFFFFu)           // tensor_dim1 hi16
                | (64u << 16));                              // tile_dim0 = 64
    g1[4] = 64;                   // tile_dim1 = 64 rows (tile_dim2 = 0)
    g1[5] = EH_;                  // tensor_dim0_stride[31:0]
    g1[6] = 0;                    // stride hi / dim1_stride lo
    g1[7] = 0;
    const v4i z4 = {0, 0, 0, 0};  // 2D tensor: groups 2/3 unused
#if __clang_major__ >= 23
    const v8i z8 = {0, 0, 0, 0, 0, 0, 0, 0};
    __builtin_amdgcn_tensor_load_to_lds(g0, g1, z4, z4, z8, 0);
#else
    __builtin_amdgcn_tensor_load_to_lds(g0, g1, z4, z4, 0);
#endif
}

// ---------------- kernel 0: weights f32 -> bf16 ----------------
__global__ void engram_cvt_kernel(const float* __restrict__ Wk,
                                  const float* __restrict__ Wv,
                                  unsigned short* __restrict__ Wkb,
                                  unsigned short* __restrict__ Wvb,
                                  int n) {
    int i = blockIdx.x * blockDim.x + threadIdx.x;
    if (i < n) {
        Wkb[i] = f2bf(Wk[i]);
        Wvb[i] = f2bf(Wv[i]);
    }
}

// ---------------- kernel 1: gather + dual GEMM (bf16 WMMA, fp32 acc) -------
// keypre[m,d] = sum_e emb[m,e]*Wk[d,e]     vpre[m,d] = sum_e emb[m,e]*Wv[d,e]
// B tiles (weights) arrive via double-buffered TDM tensor_load_to_lds.
__global__ __launch_bounds__(128)
void engram_gemm_kernel(const int*   __restrict__ hash_ids,   // [M, NH]
                        const float* __restrict__ emb_table,  // [NH, V, HD]
                        const unsigned short* __restrict__ Wkb, // [D, EH] bf16
                        const unsigned short* __restrict__ Wvb,
                        float* __restrict__ keypre,           // [M, D]
                        float* __restrict__ vpre)             // [M, D]
{
    // layout: As | Bk0 | Bv0 | Bk1 | Bv1   (each TSZ bf16 elems, 144B rows)
    __shared__ unsigned short smem[5 * TSZ];
    unsigned short* As = smem;

    const int tid  = threadIdx.x;
    const int lane = tid & 31;
    const int wave = tid >> 5;
    const int n0 = blockIdx.x * BN;
    const int m0 = blockIdx.y * BM;

    const int mhalf = (wave >> 1) & 1;   // which 32-row half of tile
    const int nhalf = wave & 1;          // which 32-col half of tile

    v8f accK[2][2] = {};
    v8f accV[2][2] = {};

    // loader roles: thread t handles LDS row t/2, 32-element half t&1
    const int lrow  = tid >> 1;
    const int lhalf = tid & 1;

    // prologue: kick off TDM for head 0 into buffer 0 (wave 0 only)
    if (wave == 0) {
        const size_t goff0 = (size_t)n0 * EH_;   // head 0 tile (col offset 0)
        tdm_load_tile_b(Wkb + goff0, (unsigned)(size_t)(smem + 1 * TSZ));
        tdm_load_tile_b(Wvb + goff0, (unsigned)(size_t)(smem + 2 * TSZ));
    }

    for (int h = 0; h < NH_; ++h) {
        const int cur = h & 1;
        const unsigned short* Bk = smem + TSZ + (2 * cur + 0) * TSZ;
        const unsigned short* Bv = smem + TSZ + (2 * cur + 1) * TSZ;

        // ---- stage A tile: gather one head's embedding rows, f32->bf16 ----
        {
            const int m  = m0 + lrow;
            const int id = hash_ids[m * NH_ + h];
            const float4* s4 =
                (const float4*)(emb_table + ((size_t)h * V_ + id) * HD_ + lhalf * 32);
            unsigned short* dst = &As[lrow * LDA + lhalf * 32];
#pragma unroll
            for (int i = 0; i < 8; ++i) {
                float4 f = s4[i];
                uint2 p;
                p.x = (unsigned)f2bf(f.x) | ((unsigned)f2bf(f.y) << 16);
                p.y = (unsigned)f2bf(f.z) | ((unsigned)f2bf(f.w) << 16);
                *(uint2*)(dst + i * 4) = p;
            }
            if (h + 1 < NH_) {   // global_prefetch_b8 of next head's gather row
                const int idn = hash_ids[m * NH_ + h + 1];
                __builtin_prefetch(
                    emb_table + ((size_t)(h + 1) * V_ + idn) * HD_ + lhalf * 32, 0, 0);
            }
        }
        // ---- TDM pipeline: issue next head's B tiles, drain current pair --
        if (wave == 0) {
            if (h + 1 < NH_) {
                const int nxt = cur ^ 1;
                const size_t goff = (size_t)n0 * EH_ + (size_t)(h + 1) * HD_;
                tdm_load_tile_b(Wkb + goff,
                                (unsigned)(size_t)(smem + TSZ + (2 * nxt + 0) * TSZ));
                tdm_load_tile_b(Wvb + goff,
                                (unsigned)(size_t)(smem + TSZ + (2 * nxt + 1) * TSZ));
                __builtin_amdgcn_s_wait_tensorcnt(2);  // current pair complete
            } else {
                __builtin_amdgcn_s_wait_tensorcnt(0);
            }
        }
        __syncthreads();

        // ---- two k-steps of 32 over this head's 64-wide K block ----
#pragma unroll
        for (int ks = 0; ks < 2; ++ks) {
            const int ko = ks * 32;
            v16bf a[2], bk[2], bv[2];
            // B fragment: lanes 0-15 -> col c, K 0..15 ; lanes 16-31 K 16..31
            const int bko = ko + ((lane >> 4) ? 16 : 0);
#pragma unroll
            for (int nt = 0; nt < 2; ++nt) {
                const int n = nhalf * 32 + nt * 16 + (lane & 15);
                FragBF fk, fv;
                fk.q[0] = *(const uint4*)&Bk[n * LDA + bko];
                fk.q[1] = *(const uint4*)&Bk[n * LDA + bko + 8];
                fv.q[0] = *(const uint4*)&Bv[n * LDA + bko];
                fv.q[1] = *(const uint4*)&Bv[n * LDA + bko + 8];
                bk[nt] = fk.v;
                bv[nt] = fv.v;
            }
            // A fragment: lane m rows; lanes 0-15 K 0..15, 16-31 K 16..31
            const int ak = ko + ((lane >> 4) ? 16 : 0);
#pragma unroll
            for (int mt = 0; mt < 2; ++mt) {
                const int r = mhalf * 32 + mt * 16 + (lane & 15);
                FragBF f;
                f.q[0] = *(const uint4*)&As[r * LDA + ak];
                f.q[1] = *(const uint4*)&As[r * LDA + ak + 8];
                a[mt] = f.v;
            }
#pragma unroll
            for (int mt = 0; mt < 2; ++mt)
#pragma unroll
                for (int nt = 0; nt < 2; ++nt) {
                    accK[mt][nt] = __builtin_amdgcn_wmma_f32_16x16x32_bf16(
                        false, a[mt], false, bk[nt], (short)0, accK[mt][nt],
                        false, false);
                    accV[mt][nt] = __builtin_amdgcn_wmma_f32_16x16x32_bf16(
                        false, a[mt], false, bv[nt], (short)0, accV[mt][nt],
                        false, false);
                }
        }
        __syncthreads();   // protect As + finished B buffer before reuse
    }

    // ---- epilogue: C/D layout lane=(n + 16*(m>=8)), vgpr v = m&7 ----
#pragma unroll
    for (int mt = 0; mt < 2; ++mt)
#pragma unroll
        for (int nt = 0; nt < 2; ++nt) {
            const int n  = n0 + nhalf * 32 + nt * 16 + (lane & 15);
            const int mb = m0 + mhalf * 32 + mt * 16 + 8 * (lane >> 4);
#pragma unroll
            for (int v = 0; v < 8; ++v) {
                const size_t idx = (size_t)(mb + v) * D_ + n;
                keypre[idx] = accK[mt][nt][v];
                vpre[idx]   = accV[mt][nt][v];
            }
        }
}

// ---------------- kernel 2: rmsnorm + gate + value -------------------------
__device__ __forceinline__ float block_reduce_sum(float x, float* red) {
#pragma unroll
    for (int o = 16; o > 0; o >>= 1) x += __shfl_xor(x, o, 32);
    const int wid  = threadIdx.x >> 5;
    const int lane = threadIdx.x & 31;
    if (lane == 0) red[wid] = x;
    __syncthreads();
    if (threadIdx.x == 0) {
        float s = 0.f;
#pragma unroll
        for (int i = 0; i < 8; ++i) s += red[i];
        red[0] = s;
    }
    __syncthreads();
    const float r = red[0];
    __syncthreads();
    return r;
}

__global__ __launch_bounds__(256)
void engram_gate_kernel(const float* __restrict__ keypre,  // [M,D]
                        const float* __restrict__ hidden,  // [M,D]
                        const float* __restrict__ bk,
                        const float* __restrict__ bv,
                        const float* __restrict__ knw,
                        const float* __restrict__ qnw,
                        float* __restrict__ vbuf)          // in: vpre, out: value
{
    __shared__ float red[8];
    const size_t m = blockIdx.x;
    const float* kp = keypre + m * D_;
    const float* hq = hidden + m * D_;

    float kreg[8], hreg[8];
    float sk = 0.f, sq = 0.f;
#pragma unroll
    for (int i = 0; i < 8; ++i) {
        const int d = threadIdx.x + i * 256;
        kreg[i] = kp[d] + bk[d];
        hreg[i] = hq[d];
        sk += kreg[i] * kreg[i];
        sq += hreg[i] * hreg[i];
    }
    sk = block_reduce_sum(sk, red);
    sq = block_reduce_sum(sq, red);
    const float eps = 1.1920929e-07f;
    const float rsk = rsqrtf(sk * (1.0f / D_) + eps);
    const float rsq = rsqrtf(sq * (1.0f / D_) + eps);

    float dot = 0.f;
#pragma unroll
    for (int i = 0; i < 8; ++i) {
        const int d = threadIdx.x + i * 256;
        dot += (kreg[i] * rsk * knw[d]) * (hreg[i] * rsq * qnw[d]);
    }
    dot = block_reduce_sum(dot, red);

    float g  = dot * 0.022097086912079608f;            // 1/sqrt(2048)
    float sg = (g > 0.f) ? 1.f : ((g < 0.f) ? -1.f : 0.f);
    float gg = sqrtf(fmaxf(fabsf(g), 1e-6f)) * sg;
    float gate = 1.f / (1.f + __expf(-gg));

#pragma unroll
    for (int i = 0; i < 8; ++i) {
        const int d = threadIdx.x + i * 256;
        const size_t idx = m * D_ + d;
        vbuf[idx] = gate * (vbuf[idx] + bv[d]);
    }
}

// ---------------- kernel 3: causal dilated depthwise conv + residual -------
__global__ __launch_bounds__(256)
void engram_conv_kernel(const float* __restrict__ value,   // [M,D]
                        const float* __restrict__ conv_w,  // [D,4]
                        const float* __restrict__ conv_b,  // [D]
                        float* __restrict__ out,
                        size_t total)
{
    const size_t i = (size_t)blockIdx.x * 256 + threadIdx.x;
    if (i >= total) return;
    const int d = (int)(i % D_);
    const int s = (int)((i / D_) % S_);

    const float x0 = value[i];
    float y = conv_b[d] + conv_w[d * 4 + 3] * x0;        // tap at s
    if (s >= 4)  y += conv_w[d * 4 + 2] * value[i - (size_t)4  * D_];
    if (s >= 8)  y += conv_w[d * 4 + 1] * value[i - (size_t)8  * D_];
    if (s >= 12) y += conv_w[d * 4 + 0] * value[i - (size_t)12 * D_];
    out[i] = x0 + y;
}

// ---------------- host launcher --------------------------------------------
extern "C" void kernel_launch(void* const* d_in, const int* in_sizes, int n_in,
                              void* d_out, int out_size, void* d_ws, size_t ws_size,
                              hipStream_t stream) {
    (void)in_sizes; (void)n_in; (void)out_size; (void)ws_size;

    const float* hidden    = (const float*)d_in[0];
    const int*   hash_ids  = (const int*)  d_in[1];
    const float* emb_table = (const float*)d_in[2];
    const float* Wk        = (const float*)d_in[3];
    const float* bk        = (const float*)d_in[4];
    const float* Wv        = (const float*)d_in[5];
    const float* bv        = (const float*)d_in[6];
    const float* knw       = (const float*)d_in[7];
    const float* qnw       = (const float*)d_in[8];
    const float* conv_w    = (const float*)d_in[9];
    const float* conv_b    = (const float*)d_in[10];
    float* out = (float*)d_out;

    // workspace layout
    char* ws = (char*)d_ws;
    const size_t wbytes = (size_t)D_ * EH_ * sizeof(unsigned short); // 6 MiB
    unsigned short* Wkb = (unsigned short*)ws;
    unsigned short* Wvb = (unsigned short*)(ws + wbytes);
    float* vbuf = (float*)(ws + 2 * wbytes);          // [M,D] 128 MiB
    float* keypre = out;                              // temp, overwritten later

    const int nw = D_ * EH_;
    engram_cvt_kernel<<<(nw + 255) / 256, 256, 0, stream>>>(Wk, Wv, Wkb, Wvb, nw);

    dim3 g1(D_ / BN, M_ / BM);  // (32, 256)
    engram_gemm_kernel<<<g1, 128, 0, stream>>>(hash_ids, emb_table, Wkb, Wvb,
                                               keypre, vbuf);

    engram_gate_kernel<<<M_, 256, 0, stream>>>(keypre, hidden, bk, bv, knw, qnw,
                                               vbuf);

    const size_t total = (size_t)M_ * D_;
    engram_conv_kernel<<<(unsigned)((total + 255) / 256), 256, 0, stream>>>(
        vbuf, conv_w, conv_b, out, total);
}
